// Attention_3GIN2_46067819217519
// MI455X (gfx1250) — matrix operand order repaired and checked by reading, not compiled, not run on hardware
//
#include <hip/hip_runtime.h>

// Problem constants (fixed by the reference)
constexpr int NB   = 4;
constexpr int NT   = 1024;
constexpr int NDIM = 256;
constexpr int NH   = 8;
constexpr int ND   = 32;    // head dim == WMMA K for bf16
constexpr int NQK  = 64;    // q|k row length in workspace

typedef __attribute__((ext_vector_type(16))) __bf16 v16bf;
typedef __attribute__((ext_vector_type(8)))  float  v8f;

// ---------- helpers ----------

__device__ inline __bf16 f2bf(float f) {
    union { float f; unsigned u; } a; a.f = f;
    unsigned r = (a.u + 0x7FFFu + ((a.u >> 16) & 1u)) >> 16;   // round-nearest-even
    union { unsigned short s; __bf16 b; } o; o.s = (unsigned short)r;
    return o.b;
}

// A-fragment (16x32 bf16) from a bf16 row: lane holds K = {16h..16h+15}
__device__ inline v16bf load_afrag_bf16(const __bf16* __restrict__ row, int kh) {
    union U { uint4 q; __bf16 b[8]; } u0, u1;
    u0.q = *(const uint4*)(row + 16 * kh);
    u1.q = *(const uint4*)(row + 16 * kh + 8);
    v16bf a;
#pragma unroll
    for (int i = 0; i < 8; ++i) { a[i] = u0.b[i]; a[8 + i] = u1.b[i]; }
    return a;
}

// A-fragment (16x32 bf16) from an fp32 row (only used for attn probabilities)
__device__ inline v16bf load_afrag_f32(const float* __restrict__ row, int kh) {
    const float* r1 = row + 16 * kh;
    const float* r2 = row + 16 * kh + 8;
    float4 a0 = *(const float4*)r1;
    float4 a1 = *(const float4*)(r1 + 4);
    float4 c0 = *(const float4*)r2;
    float4 c1 = *(const float4*)(r2 + 4);
    v16bf a;
    a[0]=f2bf(a0.x); a[1]=f2bf(a0.y); a[2]=f2bf(a0.z); a[3]=f2bf(a0.w);
    a[4]=f2bf(a1.x); a[5]=f2bf(a1.y); a[6]=f2bf(a1.z); a[7]=f2bf(a1.w);
    a[8]=f2bf(c0.x); a[9]=f2bf(c0.y); a[10]=f2bf(c0.z); a[11]=f2bf(c0.w);
    a[12]=f2bf(c1.x); a[13]=f2bf(c1.y); a[14]=f2bf(c1.z); a[15]=f2bf(c1.w);
    return a;
}

// B-fragment (32x16 bf16): this lane's 16 K-values contiguous in memory
__device__ inline v16bf load_bfrag_contig(const __bf16* __restrict__ p) {
    union U { uint4 q; __bf16 b[8]; } u0, u1;
    u0.q = *(const uint4*)p;
    u1.q = *(const uint4*)(p + 8);
    v16bf r;
#pragma unroll
    for (int i = 0; i < 8; ++i) { r[i] = u0.b[i]; r[8 + i] = u1.b[i]; }
    return r;
}

#define WMMA_BF16(A, B, C) \
    __builtin_amdgcn_wmma_f32_16x16x32_bf16(false, (A), false, (B), (short)0, (C), false, false)

// ---------- K0a: adj fp32 -> bf16 (read once at half bandwidth afterwards) ----------
__global__ __launch_bounds__(256) void k_cvt_adj(
    const float* __restrict__ in, unsigned short* __restrict__ out_raw)
{
    __bf16* out = reinterpret_cast<__bf16*>(out_raw);
    const size_t i = ((size_t)blockIdx.x * 256 + threadIdx.x) * 8;
    float4 a = *(const float4*)(in + i);
    float4 b = *(const float4*)(in + i + 4);
    union { uint4 q; __bf16 h[8]; } u;
    u.h[0]=f2bf(a.x); u.h[1]=f2bf(a.y); u.h[2]=f2bf(a.z); u.h[3]=f2bf(a.w);
    u.h[4]=f2bf(b.x); u.h[5]=f2bf(b.y); u.h[6]=f2bf(b.z); u.h[7]=f2bf(b.w);
    *(uint4*)(out + i) = u.q;
}

// ---------- K0b: xT[b][c][t] = bf16(x[b][t][c])  (B-fragments become contiguous) ----------
__global__ __launch_bounds__(256) void k_cvt_xT(
    const float* __restrict__ x, unsigned short* __restrict__ out_raw)
{
    __bf16* xT = reinterpret_cast<__bf16*>(out_raw);
    const int tid = blockIdx.x * 256 + threadIdx.x;      // 0 .. 4*256*256-1
    const int tq = tid & 255;                            // t/4
    const int c  = (tid >> 8) & 255;
    const int b  = tid >> 16;
    const int t  = tq * 4;
    union { uint2 q; __bf16 h[4]; } u;
#pragma unroll
    for (int j = 0; j < 4; ++j)
        u.h[j] = f2bf(x[((size_t)(b * NT + t + j)) * NDIM + c]);
    *(uint2*)(xT + ((size_t)(b * NDIM + c)) * NT + t) = u.q;
}

// ---------- K1: agg = (1+eps)*x + adj@x ; q,k -> qk ws (q pre-scaled), v -> vT ws ----------
// 2048 waves: (b, h, mtile). Each wave: 16 rows x 32 cols (one head).
__global__ __launch_bounds__(256) void k_gin_qkv(
    const float* __restrict__ x, const unsigned short* __restrict__ adjb_raw,
    const unsigned short* __restrict__ xT_raw,
    const float* __restrict__ W, const float* __restrict__ eps,
    unsigned short* __restrict__ qk_raw, unsigned short* __restrict__ vT_raw)
{
    const __bf16* adjb = reinterpret_cast<const __bf16*>(adjb_raw);
    const __bf16* xT   = reinterpret_cast<const __bf16*>(xT_raw);
    __bf16* qk = reinterpret_cast<__bf16*>(qk_raw);
    __bf16* vT = reinterpret_cast<__bf16*>(vT_raw);

    const int lane = threadIdx.x & 31;
    const int wave = threadIdx.x >> 5;
    const int wid  = blockIdx.x * 8 + wave;        // 0..2047
    const int mtile = wid & 63;
    const int h     = (wid >> 6) & 7;
    const int b     = wid >> 9;
    const int kh  = lane >> 4;
    const int l16 = lane & 15;
    const int tm  = mtile * 16;
    const int bh  = b * NH + h;

    const __bf16* arow = adjb + (size_t)(b * NT + tm + l16) * NT;      // A: adj row
    const __bf16* xc0  = xT + (size_t)(b * NDIM + h * ND + l16) * NT;  // B: x col (head)
    const __bf16* xc1  = xc0 + (size_t)16 * NT;
    const float*  x_b  = x + (size_t)b * NT * NDIM;

    v8f acc0 = {}; v8f acc1 = {};
    for (int kk = 0; kk < NT; kk += 32) {
        __builtin_prefetch(arow + kk + 64, 0, 0);      // next-next A tile
        v16bf a  = load_afrag_bf16(arow + kk, kh);
        v16bf b0 = load_bfrag_contig(xc0 + kk + 16 * kh);
        v16bf b1 = load_bfrag_contig(xc1 + kk + 16 * kh);
        acc0 = WMMA_BF16(a, b0, acc0);
        acc1 = WMMA_BF16(a, b1, acc1);
    }

    const float e1 = 1.0f + eps[0];
#pragma unroll
    for (int r = 0; r < 8; ++r) {
        const int row = tm + r + 8 * kh;
        acc0[r] += e1 * x_b[(size_t)row * NDIM + h * ND + l16];
        acc1[r] += e1 * x_b[(size_t)row * NDIM + h * ND + 16 + l16];
    }

    // C-layout -> A-layout relayout via LDS (16x32 f32 tile per wave)
    __shared__ float lds[8][16 * 32];
    float* tl = lds[wave];
#pragma unroll
    for (int r = 0; r < 8; ++r) {
        tl[(r + 8 * kh) * 32 + l16]      = acc0[r];
        tl[(r + 8 * kh) * 32 + 16 + l16] = acc1[r];
    }
    __syncthreads();

    v16bf ag;
    {
        const float* row = tl + l16 * 32;
#pragma unroll
        for (int i = 0; i < 8; ++i) {
            ag[i]     = f2bf(row[16 * kh + i]);
            ag[8 + i] = f2bf(row[16 * kh + 8 + i]);
        }
    }

    // agg(16x32) @ W(32x96): j=0..3 -> q|k (q scaled by dim^-0.5 = 1/16), j=4..5 -> vT
    __bf16* qk_bh = qk + (size_t)bh * NT * NQK;
#pragma unroll
    for (int j = 0; j < 6; ++j) {
        v16bf wb;
#pragma unroll
        for (int i = 0; i < 16; ++i)
            wb[i] = f2bf(W[(kh * 16 + i) * 96 + j * 16 + l16]);
        v8f dz = {};
        dz = WMMA_BF16(ag, wb, dz);
        if (j < 4) {
            const float mul = (j < 2) ? 0.0625f : 1.0f;
            const int colq = j * 16 + l16;
#pragma unroll
            for (int r = 0; r < 8; ++r)
                qk_bh[(size_t)(tm + r + 8 * kh) * NQK + colq] = f2bf(dz[r] * mul);
        } else {
            // vT[bh][colv][t]: lane's 8 rows are contiguous in t -> one 16B store
            const int colv = (j - 4) * 16 + l16;
            union { uint4 q; __bf16 hh[8]; } u;
#pragma unroll
            for (int r = 0; r < 8; ++r) u.hh[r] = f2bf(dz[r]);
            *(uint4*)(vT + ((size_t)(bh * ND + colv)) * NT + tm + 8 * kh) = u.q;
        }
    }
}

// ---------- K2: logits S = (q/16) . k^T  (one WMMA per 16x16 tile, K=32=d) ----------
// 8192 waves: (bh, mtile, strip of 256 cols)
__global__ __launch_bounds__(256) void k_logits(
    const unsigned short* __restrict__ qk_raw, float* __restrict__ attn)
{
    const __bf16* qk = reinterpret_cast<const __bf16*>(qk_raw);
    const int lane = threadIdx.x & 31;
    const int wave = threadIdx.x >> 5;
    const int wid  = blockIdx.x * 8 + wave;        // 0..8191
    const int strip = wid & 3;
    const int mtile = (wid >> 2) & 63;
    const int bh    = wid >> 8;
    const int kh  = lane >> 4;
    const int l16 = lane & 15;
    const int tm  = mtile * 16;

    const __bf16* qk_bh = qk + (size_t)bh * NT * NQK;
    float* attn_bh = attn + (size_t)bh * NT * NT;

    // q-tile A-fragment, reused across the whole 256-wide strip
    v16bf a = load_afrag_bf16(qk_bh + (size_t)(tm + l16) * NQK, kh);

#pragma unroll 1
    for (int j = 0; j < 16; ++j) {
        const int ncol = strip * 256 + j * 16;
        // B[k][n] = K[ncol+n][k]; lane's 16 K-dims contiguous at offset 32 (k block)
        v16bf bb = load_bfrag_contig(qk_bh + (size_t)(ncol + l16) * NQK + 32 + 16 * kh);
        v8f dz = {};
        dz = WMMA_BF16(a, bb, dz);
#pragma unroll
        for (int r = 0; r < 8; ++r)
            attn_bh[(size_t)(tm + r + 8 * kh) * NT + ncol + l16] = dz[r];
    }
}

// ---------- K3: row softmax over 1024 (in place, L2-resident) ----------
__global__ __launch_bounds__(256) void k_softmax(float* __restrict__ attn)
{
    float* p = attn + (size_t)blockIdx.x * NT;
    const int tid = threadIdx.x;
    float v[4];
    float mx = -3.4e38f;
#pragma unroll
    for (int i = 0; i < 4; ++i) { v[i] = p[tid + i * 256]; mx = fmaxf(mx, v[i]); }
    __shared__ float red[256];
    red[tid] = mx; __syncthreads();
    for (int s = 128; s > 0; s >>= 1) {
        if (tid < s) red[tid] = fmaxf(red[tid], red[tid + s]);
        __syncthreads();
    }
    mx = red[0]; __syncthreads();
    float sum = 0.0f;
#pragma unroll
    for (int i = 0; i < 4; ++i) { v[i] = __expf(v[i] - mx); sum += v[i]; }
    red[tid] = sum; __syncthreads();
    for (int s = 128; s > 0; s >>= 1) {
        if (tid < s) red[tid] += red[tid + s];
        __syncthreads();
    }
    const float inv = 1.0f / red[0];
#pragma unroll
    for (int i = 0; i < 4; ++i) p[tid + i * 256] = v[i] * inv;
}

// ---------- K4: O = P@V, exact GELU, store b,t,(h d) ----------
// 2048 waves: (bh, mtile), each 16 rows x 32 cols
__global__ __launch_bounds__(256) void k_av_gelu(
    const float* __restrict__ attn, const unsigned short* __restrict__ vT_raw,
    float* __restrict__ out)
{
    const __bf16* vT = reinterpret_cast<const __bf16*>(vT_raw);
    const int lane = threadIdx.x & 31;
    const int wave = threadIdx.x >> 5;
    const int wid  = blockIdx.x * 8 + wave;        // 0..2047
    const int mtile = wid & 63;
    const int bh    = wid >> 6;
    const int b = bh >> 3, h = bh & 7;
    const int kh  = lane >> 4;
    const int l16 = lane & 15;
    const int tm  = mtile * 16;

    const float*  prow = attn + (size_t)(bh * NT + tm + l16) * NT;
    const __bf16* vc0  = vT + ((size_t)(bh * ND + l16)) * NT;       // B col (dims 0..15)
    const __bf16* vc1  = vc0 + (size_t)16 * NT;                     // dims 16..31

    v8f acc0 = {}, acc1 = {};
    for (int kk = 0; kk < NT; kk += 32) {
        v16bf a  = load_afrag_f32(prow + kk, kh);
        v16bf b0 = load_bfrag_contig(vc0 + kk + 16 * kh);
        v16bf b1 = load_bfrag_contig(vc1 + kk + 16 * kh);
        acc0 = WMMA_BF16(a, b0, acc0);
        acc1 = WMMA_BF16(a, b1, acc1);
    }

#pragma unroll
    for (int r = 0; r < 8; ++r) {
        const int row = tm + r + 8 * kh;
        float o0 = acc0[r], o1 = acc1[r];
        o0 = 0.5f * o0 * (1.0f + erff(o0 * 0.70710678118654752f));
        o1 = 0.5f * o1 * (1.0f + erff(o1 * 0.70710678118654752f));
        float* orow = out + ((size_t)b * NT + row) * NDIM + h * ND;
        orow[l16]      = o0;
        orow[16 + l16] = o1;
    }
}

// ---------- launch ----------
extern "C" void kernel_launch(void* const* d_in, const int* in_sizes, int n_in,
                              void* d_out, int out_size, void* d_ws, size_t ws_size,
                              hipStream_t stream) {
    const float* x   = (const float*)d_in[0];
    const float* adj = (const float*)d_in[1];
    // d_in[2] = rep_adj_dis: unused in forward
    const float* W   = (const float*)d_in[3];
    const float* eps = (const float*)d_in[4];

    float* out  = (float*)d_out;                       // [4,1024,256]
    float* attn = out + (size_t)NB * NT * NDIM;        // [4,8,1024,1024]

    // workspace: bf16 buffers (16 MB total)
    unsigned short* qk   = (unsigned short*)d_ws;                 // [32][1024][64]
    unsigned short* vT   = qk   + (size_t)32 * NT * NQK;          // [32][32][1024]
    unsigned short* adjb = vT   + (size_t)32 * ND * NT;           // [4][1024][1024]
    unsigned short* xT   = adjb + (size_t)NB * NT * NT;           // [4][256][1024]

    k_cvt_adj <<<2048, 256, 0, stream>>>(adj, adjb);
    k_cvt_xT  <<<1024, 256, 0, stream>>>(x, xT);
    k_gin_qkv <<<256,  256, 0, stream>>>(x, adjb, xT, W, eps, qk, vT);
    k_logits  <<<1024, 256, 0, stream>>>(qk, attn);
    k_softmax <<<NB * NH * NT, 256, 0, stream>>>(attn);
    k_av_gelu <<<256,  256, 0, stream>>>(attn, vT, out);
}
